// Attention_24326694764832
// MI455X (gfx1250) — matrix operand order repaired and checked by reading, not compiled, not run on hardware
//
#include <hip/hip_runtime.h>
#include <hip/hip_bf16.h>
#include <math.h>

// ---------------------------------------------------------------------------
// LinearAttention (16,512,64,64) for MI455X / gfx1250, wave32, WMMA bf16.
// Bandwidth-bound (~1 GB min traffic @ 23.3 TB/s ~= 45 us; 36.5 GFLOP is
// negligible on the WMMA units) -> all GEMMs via v_wmma_f32_16x16x32_bf16,
// fp32->bf16 packed conversion during LDS staging, register double-buffering
// so global_load_b64s are in flight while WMMAs execute.
// ---------------------------------------------------------------------------

typedef __bf16 bf16_t;
typedef bf16_t v16bf __attribute__((ext_vector_type(16)));
typedef bf16_t v8bf  __attribute__((ext_vector_type(8)));
typedef bf16_t v2bf  __attribute__((ext_vector_type(2)));
typedef float  v8f   __attribute__((ext_vector_type(8)));

#define LDA 40  // padded LDS row stride (elements); keeps 8-elem groups 16B aligned

// pack two fp32 -> one dword of bf16x2 (native fptrunc, RNE)
__device__ __forceinline__ unsigned pack2(float a, float b) {
  union { v2bf v; unsigned u; } c;
  c.v[0] = (bf16_t)a;
  c.v[1] = (bf16_t)b;
  return c.u;
}

// A-matrix 16x32 bf16 fragment (ISA 7.12.2): lane half selects K-octet pairing.
__device__ __forceinline__ v16bf load_frag_A(const bf16_t* row, int lane) {
  const int kb = ((lane >> 4) & 1) * 8;
  v8bf lo = *reinterpret_cast<const v8bf*>(row + kb);        // K = kb..kb+7
  v8bf hi = *reinterpret_cast<const v8bf*>(row + 16 + kb);   // K = 16+kb..
  return __builtin_shufflevector(lo, hi, 0,1,2,3,4,5,6,7,8,9,10,11,12,13,14,15);
}

// B-matrix 32x16 bf16 fragment: B staged N-major, lane (col=lane&15) reads
// K = kb..kb+7 and 16+kb..16+kb+7; lane half selects kb = 0 or 8 (matches A).
__device__ __forceinline__ v16bf load_frag_B(const bf16_t* row, int lane) {
  const int kb = ((lane >> 4) & 1) * 8;
  v8bf lo = *reinterpret_cast<const v8bf*>(row + kb);
  v8bf hi = *reinterpret_cast<const v8bf*>(row + 16 + kb);
  return __builtin_shufflevector(lo, hi, 0,1,2,3,4,5,6,7,8,9,10,11,12,13,14,15);
}

__device__ __forceinline__ v8f wmma_bf16(v16bf a, v16bf b, v8f c) {
  return __builtin_amdgcn_wmma_f32_16x16x32_bf16(false, a, false, b,
                                                 (short)0, c, false, false);
}

// ---------------------------------------------------------------------------
// Kernel 1: qkv[b] = W_qkv(384x512) * x[b](512x4096).  Tile 128x128xK32,
// 256 threads = 8 waves (2x4), each wave 64x32 = 4x2 WMMA tiles.
// Register double-buffered staging: next panel's loads issue before the WMMAs.
// ---------------------------------------------------------------------------
__global__ void __launch_bounds__(256) qkv_gemm(const float* __restrict__ x,
                                                const float* __restrict__ Wqkv,
                                                float* __restrict__ qkv) {
  __shared__ bf16_t As[128][LDA];
  __shared__ bf16_t Bs[128][LDA];
  const int tid  = threadIdx.x;
  const int lane = tid & 31;
  const int wid  = tid >> 5;
  const int wm   = wid >> 2;   // 0..1 (64 rows)
  const int wn   = wid & 3;    // 0..3 (32 cols)
  const int n0   = blockIdx.x * 128;
  const int m0   = blockIdx.y * 128;
  const int b    = blockIdx.z;
  const float* X = x + (size_t)b * 512 * 4096;

  // per-thread staging slots: 8 A pairs (K-contig) + 8 B pairs (K-contig)
  const int ar_r = tid >> 4;            // A row of slot e=0, 0..15 (e-slots 16 rows apart)
  float2 ar[8];                         // A: pairs along K
  float  bra[8], brb[8];                // B: two K-rows, same n

  auto load_panel = [&](int k0) {
#pragma unroll
    for (int e = 0; e < 8; ++e) {
      int p = tid + e * 256;            // 0..2047 pairs
      int r = p >> 4, c = (p & 15) * 2; // A[r][c..c+1]
      ar[e] = *reinterpret_cast<const float2*>(
          &Wqkv[(size_t)(m0 + r) * 512 + k0 + c]);
      int kk = (p >> 7) * 2, nn = p & 127;  // B[kk..kk+1][nn]
      const float* src = X + (size_t)(k0 + kk) * 4096 + n0 + nn;
      bra[e] = src[0];
      brb[e] = src[4096];
    }
  };
  auto store_panel = [&]() {
#pragma unroll
    for (int e = 0; e < 8; ++e) {
      int p = tid + e * 256;
      int r = p >> 4, c = (p & 15) * 2;
      *reinterpret_cast<unsigned*>(&As[r][c]) = pack2(ar[e].x, ar[e].y);
      int kk = (p >> 7) * 2, nn = p & 127;
      *reinterpret_cast<unsigned*>(&Bs[nn][kk]) = pack2(bra[e], brb[e]);
    }
  };

  const v8f vzero = {0.f,0.f,0.f,0.f,0.f,0.f,0.f,0.f};
  v8f acc[4][2];
#pragma unroll
  for (int i = 0; i < 4; ++i)
#pragma unroll
    for (int j = 0; j < 2; ++j) acc[i][j] = vzero;

  load_panel(0);
  for (int k0 = 0; k0 < 512; k0 += 32) {
    store_panel();
    __syncthreads();
    if (k0 + 32 < 512) load_panel(k0 + 32);   // in flight during WMMAs

    v16bf bfrag[2];
#pragma unroll
    for (int nt = 0; nt < 2; ++nt)
      bfrag[nt] = load_frag_B(&Bs[wn * 32 + nt * 16 + (lane & 15)][0], lane);
#pragma unroll
    for (int mt = 0; mt < 4; ++mt) {
      v16bf afrag = load_frag_A(&As[wm * 64 + mt * 16 + (lane & 15)][0], lane);
#pragma unroll
      for (int nt = 0; nt < 2; ++nt)
        acc[mt][nt] = wmma_bf16(afrag, bfrag[nt], acc[mt][nt]);
    }
    __syncthreads();
  }
  (void)ar_r;

  // C layout (ISA): VGPR v -> row v + 8*(lane>>4); col = lane&15
  const int rsel = (lane >> 4) * 8;
  const int col  = lane & 15;
#pragma unroll
  for (int mt = 0; mt < 4; ++mt)
#pragma unroll
    for (int nt = 0; nt < 2; ++nt)
#pragma unroll
      for (int v = 0; v < 8; ++v) {
        int m = m0 + wm * 64 + mt * 16 + rsel + v;
        int n = n0 + wn * 32 + nt * 16 + col;
        qkv[((size_t)b * 384 + m) * 4096 + n] = acc[mt][nt][v];
      }
}

// ---------------------------------------------------------------------------
// Kernel 2: per (b,h): context[d][e] = sum_n softmax_n(k)[d][n] * v[e][n].
// Phase 1: online softmax stats over n=4096 (wave32 shfl reduce).
// Phase 2: K-accumulated 32x32 WMMA (2x2 tiles, one per wave), 128 K-steps.
// ---------------------------------------------------------------------------
__global__ void __launch_bounds__(128) ctx_kernel(const float* __restrict__ qkv,
                                                  float* __restrict__ ctx) {
  const int b = blockIdx.x >> 2, h = blockIdx.x & 3;
  const float* Kp = qkv + ((size_t)b * 384 + 128 + h * 32) * 4096;
  const float* Vp = qkv + ((size_t)b * 384 + 256 + h * 32) * 4096;
  const int tid = threadIdx.x, lane = tid & 31, wid = tid >> 5;

  __shared__ float  rowmax[32], rowinv[32];
  __shared__ bf16_t ks[32][LDA];
  __shared__ bf16_t vs[32][LDA];

  // phase 1: each wave owns 8 rows of k
  for (int dd = 0; dd < 8; ++dd) {
    int d = wid * 8 + dd;
    float m = -3.4e38f, s = 0.f;
    for (int n = lane; n < 4096; n += 32) {
      float v = Kp[(size_t)d * 4096 + n];
      float nm = fmaxf(m, v);
      s = s * __expf(m - nm) + __expf(v - nm);
      m = nm;
    }
#pragma unroll
    for (int off = 16; off > 0; off >>= 1) {
      float om = __shfl_xor(m, off);
      float os = __shfl_xor(s, off);
      float nm = fmaxf(m, om);
      s = s * __expf(m - nm) + os * __expf(om - nm);
      m = nm;
    }
    if (lane == 0) { rowmax[d] = m; rowinv[d] = 1.f / s; }
  }
  __syncthreads();

  const int mt = wid >> 1, nt = wid & 1;   // wave -> 16x16 tile of 32x32 out
  const v8f vzero = {0.f,0.f,0.f,0.f,0.f,0.f,0.f,0.f};
  v8f acc = vzero;

  for (int n0 = 0; n0 < 4096; n0 += 32) {
#pragma unroll
    for (int e = 0; e < 4; ++e) {
      int p = tid + e * 128;            // 0..511 pairs
      int r = p >> 4, c = (p & 15) * 2; // row r, cols c..c+1 (K-contig)
      const float* kp = Kp + (size_t)r * 4096 + n0 + c;
      const float* vp = Vp + (size_t)r * 4096 + n0 + c;
      float k0f = __expf(kp[0] - rowmax[r]) * rowinv[r];
      float k1f = __expf(kp[1] - rowmax[r]) * rowinv[r];
      *reinterpret_cast<unsigned*>(&ks[r][c]) = pack2(k0f, k1f);
      *reinterpret_cast<unsigned*>(&vs[r][c]) = pack2(vp[0], vp[1]);
    }
    __syncthreads();
    v16bf a  = load_frag_A(&ks[mt * 16 + (lane & 15)][0], lane);
    v16bf bb = load_frag_B(&vs[nt * 16 + (lane & 15)][0], lane);  // B[n][e]=v[e][n]
    acc = wmma_bf16(a, bb, acc);
    __syncthreads();
  }

  const int rsel = (lane >> 4) * 8, col = lane & 15;
#pragma unroll
  for (int v = 0; v < 8; ++v) {
    int d = mt * 16 + rsel + v, e = nt * 16 + col;
    ctx[(((size_t)b * 4 + h) * 32 + d) * 32 + e] = acc[v];
  }
}

// ---------------------------------------------------------------------------
// Kernel 3: per (b,h,n-tile of 128): out[e][n] = sum_d ctx[d][e] * q'[d][n]
// with q' = scale * softmax_d(q).  K=32 -> one WMMA per 16x16 tile.
// ---------------------------------------------------------------------------
__global__ void __launch_bounds__(128) attn_apply(const float* __restrict__ qkv,
                                                  const float* __restrict__ ctx,
                                                  float* __restrict__ attn) {
  const int n0 = blockIdx.x * 128;
  const int h = blockIdx.y, b = blockIdx.z;
  const float* Qp = qkv + ((size_t)b * 384 + h * 32) * 4096;
  const float* Cp = ctx + (((size_t)b * 4 + h) * 32) * 32;
  const int tid = threadIdx.x, lane = tid & 31, wid = tid >> 5;

  __shared__ bf16_t ct[32][LDA];     // ct[e][d] = ctx[d][e]  (A, M-major)
  __shared__ float  qraw[32][132];   // raw q tile, padded
  __shared__ bf16_t qs[128][LDA];    // qs[n][d] = q'[d][n]   (B, N-major)

#pragma unroll
  for (int e = 0; e < 4; ++e) {
    int p = tid + e * 128;            // 0..511 pairs
    int r = p >> 4, c = (p & 15) * 2; // ct[r][c..c+1]
    *reinterpret_cast<unsigned*>(&ct[r][c]) =
        pack2(Cp[c * 32 + r], Cp[(c + 1) * 32 + r]);
  }
#pragma unroll
  for (int d = 0; d < 32; ++d)
    qraw[d][tid] = Qp[(size_t)d * 4096 + n0 + tid];
  __syncthreads();

  {  // per-column softmax over d (thread = column)
    float m = -3.4e38f;
#pragma unroll
    for (int d = 0; d < 32; ++d) m = fmaxf(m, qraw[d][tid]);
    float s = 0.f;
#pragma unroll
    for (int d = 0; d < 32; ++d) s += __expf(qraw[d][tid] - m);
    float r = 0.17677669529663689f / s;   // scale = 32^-0.5
#pragma unroll
    for (int d = 0; d < 32; d += 2) {
      float e0 = __expf(qraw[d][tid] - m) * r;
      float e1 = __expf(qraw[d + 1][tid] - m) * r;
      *reinterpret_cast<unsigned*>(&qs[tid][d]) = pack2(e0, e1);
    }
  }
  __syncthreads();

  const v8f vzero = {0.f,0.f,0.f,0.f,0.f,0.f,0.f,0.f};
  v8f acc[2][2];
#pragma unroll
  for (int mt = 0; mt < 2; ++mt) {
    v16bf a = load_frag_A(&ct[mt * 16 + (lane & 15)][0], lane);
#pragma unroll
    for (int nt = 0; nt < 2; ++nt) {
      v16bf bb = load_frag_B(&qs[wid * 32 + nt * 16 + (lane & 15)][0], lane);
      acc[mt][nt] = wmma_bf16(a, bb, vzero);
    }
  }

  const int rsel = (lane >> 4) * 8, col = lane & 15;
#pragma unroll
  for (int mt = 0; mt < 2; ++mt)
#pragma unroll
    for (int nt = 0; nt < 2; ++nt)
#pragma unroll
      for (int v = 0; v < 8; ++v) {
        int e = mt * 16 + rsel + v;
        int n = n0 + wid * 32 + nt * 16 + col;
        attn[((size_t)b * 128 + h * 32 + e) * 4096 + n] = acc[mt][nt][v];
      }
}

// ---------------------------------------------------------------------------
// Kernel 4: out[b] = W_out(512x128) * attn[b](128x4096) + b_out, plus fused
// per-batch sum/sumsq accumulation (GroupNorm stats) via block reduce+atomics.
// ---------------------------------------------------------------------------
__global__ void __launch_bounds__(256) out_gemm(const float* __restrict__ attn,
                                                const float* __restrict__ Wout,
                                                const float* __restrict__ bout,
                                                float* __restrict__ out,
                                                float* __restrict__ stats) {
  __shared__ bf16_t As[128][LDA];
  __shared__ bf16_t Bs[128][LDA];
  __shared__ float  red[256];
  const int tid  = threadIdx.x;
  const int lane = tid & 31;
  const int wid  = tid >> 5;
  const int wm   = wid >> 2;
  const int wn   = wid & 3;
  const int n0   = blockIdx.x * 128;
  const int m0   = blockIdx.y * 128;
  const int b    = blockIdx.z;
  const float* Bp = attn + (size_t)b * 128 * 4096;

  float2 ar[8];
  float  bra[8], brb[8];
  auto load_panel = [&](int k0) {
#pragma unroll
    for (int e = 0; e < 8; ++e) {
      int p = tid + e * 256;
      int r = p >> 4, c = (p & 15) * 2;
      ar[e] = *reinterpret_cast<const float2*>(
          &Wout[(size_t)(m0 + r) * 128 + k0 + c]);
      int kk = (p >> 7) * 2, nn = p & 127;
      const float* src = Bp + (size_t)(k0 + kk) * 4096 + n0 + nn;
      bra[e] = src[0];
      brb[e] = src[4096];
    }
  };
  auto store_panel = [&]() {
#pragma unroll
    for (int e = 0; e < 8; ++e) {
      int p = tid + e * 256;
      int r = p >> 4, c = (p & 15) * 2;
      *reinterpret_cast<unsigned*>(&As[r][c]) = pack2(ar[e].x, ar[e].y);
      int kk = (p >> 7) * 2, nn = p & 127;
      *reinterpret_cast<unsigned*>(&Bs[nn][kk]) = pack2(bra[e], brb[e]);
    }
  };

  const v8f vzero = {0.f,0.f,0.f,0.f,0.f,0.f,0.f,0.f};
  v8f acc[4][2];
#pragma unroll
  for (int i = 0; i < 4; ++i)
#pragma unroll
    for (int j = 0; j < 2; ++j) acc[i][j] = vzero;

  load_panel(0);
  for (int k0 = 0; k0 < 128; k0 += 32) {
    store_panel();
    __syncthreads();
    if (k0 + 32 < 128) load_panel(k0 + 32);

    v16bf bfrag[2];
#pragma unroll
    for (int nt = 0; nt < 2; ++nt)
      bfrag[nt] = load_frag_B(&Bs[wn * 32 + nt * 16 + (lane & 15)][0], lane);
#pragma unroll
    for (int mt = 0; mt < 4; ++mt) {
      v16bf afrag = load_frag_A(&As[wm * 64 + mt * 16 + (lane & 15)][0], lane);
#pragma unroll
      for (int nt = 0; nt < 2; ++nt)
        acc[mt][nt] = wmma_bf16(afrag, bfrag[nt], acc[mt][nt]);
    }
    __syncthreads();
  }

  const int rsel = (lane >> 4) * 8;
  const int col  = lane & 15;
  float lsum = 0.f, lsq = 0.f;
#pragma unroll
  for (int mt = 0; mt < 4; ++mt)
#pragma unroll
    for (int nt = 0; nt < 2; ++nt)
#pragma unroll
      for (int v = 0; v < 8; ++v) {
        int m = m0 + wm * 64 + mt * 16 + rsel + v;
        int n = n0 + wn * 32 + nt * 16 + col;
        float y = acc[mt][nt][v] + bout[m];
        out[((size_t)b * 512 + m) * 4096 + n] = y;
        lsum += y;
        lsq  += y * y;
      }

  red[tid] = lsum;
  __syncthreads();
  for (int s = 128; s > 0; s >>= 1) {
    if (tid < s) red[tid] += red[tid + s];
    __syncthreads();
  }
  if (tid == 0) atomicAdd(&stats[2 * b], red[0]);
  __syncthreads();
  red[tid] = lsq;
  __syncthreads();
  for (int s = 128; s > 0; s >>= 1) {
    if (tid < s) red[tid] += red[tid + s];
    __syncthreads();
  }
  if (tid == 0) atomicAdd(&stats[2 * b + 1], red[0]);
}

// ---------------------------------------------------------------------------
// Kernel 5: GroupNorm(1 group) apply:  out = (y-mean)*rsqrt(var+eps)*g + b
// ---------------------------------------------------------------------------
__global__ void __launch_bounds__(256) gn_apply(float* __restrict__ out,
                                                const float* __restrict__ stats,
                                                const float* __restrict__ gamma,
                                                const float* __restrict__ beta) {
  size_t idx = (size_t)blockIdx.x * 256 + threadIdx.x;
  int b = (int)(idx >> 21);            // 512*4096 = 2^21 elements per batch
  int c = (int)((idx >> 12) & 511);    // 4096 = 2^12 per channel
  const float invN = 1.f / 2097152.f;
  float mean = stats[2 * b] * invN;
  float var  = stats[2 * b + 1] * invN - mean * mean;
  float inv  = rsqrtf(var + 1e-5f);
  float y = out[idx];
  out[idx] = (y - mean) * inv * gamma[c] + beta[c];
}

__global__ void zero_stats(float* __restrict__ stats) {
  if (threadIdx.x < 32) stats[threadIdx.x] = 0.f;
}

// ---------------------------------------------------------------------------
extern "C" void kernel_launch(void* const* d_in, const int* in_sizes, int n_in,
                              void* d_out, int out_size, void* d_ws, size_t ws_size,
                              hipStream_t stream) {
  const float* x     = (const float*)d_in[0];
  const float* Wqkv  = (const float*)d_in[1];
  const float* Wout  = (const float*)d_in[2];
  const float* bout  = (const float*)d_in[3];
  const float* gamma = (const float*)d_in[4];
  const float* beta  = (const float*)d_in[5];
  float* out = (float*)d_out;

  float* ws    = (float*)d_ws;
  float* qkv   = ws;                      // 16*384*4096  = 25165824 floats
  float* attn  = qkv + 25165824;          // 16*128*4096  =  8388608 floats
  float* ctx   = attn + 8388608;          // 16*4*32*32   =    65536 floats
  float* stats = ctx + 65536;             // 32 floats

  zero_stats<<<1, 32, 0, stream>>>(stats);
  qkv_gemm  <<<dim3(32, 3, 16), 256, 0, stream>>>(x, Wqkv, qkv);
  ctx_kernel<<<64, 128, 0, stream>>>(qkv, ctx);
  attn_apply<<<dim3(32, 4, 16), 128, 0, stream>>>(qkv, ctx, attn);
  out_gemm  <<<dim3(32, 4, 16), 256, 0, stream>>>(attn, Wout, bout, out, stats);
  gn_apply  <<<131072, 256, 0, stream>>>(out, stats, gamma, beta);
}
